// MultiHeadSelfAttention_34548716929384
// MI455X (gfx1250) — matrix operand order repaired and checked by reading, not compiled, not run on hardware
//
#include <hip/hip_runtime.h>

#define B_  8
#define S_  1024
#define H_  768
#define NH_ 12
#define HD_ 64
#define M_  (B_ * S_)   // 8192

typedef __bf16 bf16;
typedef __attribute__((ext_vector_type(16))) __bf16 v16bf;
typedef __attribute__((ext_vector_type(8)))  __bf16 v8bf;
typedef __attribute__((ext_vector_type(2)))  __bf16 v2bf;
typedef __attribute__((ext_vector_type(8)))  float  v8f;
typedef __attribute__((ext_vector_type(2)))  float  v2f;
typedef __attribute__((ext_vector_type(4)))  unsigned int v4u;
typedef __attribute__((ext_vector_type(4)))  int v4i;
typedef __attribute__((ext_vector_type(8)))  int v8i;

#if __has_builtin(__builtin_amdgcn_tensor_load_to_lds) && \
    __has_builtin(__builtin_amdgcn_s_wait_tensorcnt)
#define HAVE_TDM 1
#else
#define HAVE_TDM 0
#endif

// ---------- conversion helpers ----------
// Vector fptrunc: lets ISel select packed v_cvt_pk_bf16_f32 on gfx1250.

static __device__ __forceinline__ v2bf pk2(float a, float b) {
  v2f t; t[0] = a; t[1] = b;
  return __builtin_convertvector(t, v2bf);
}

static __device__ __forceinline__ bf16 f2bf1(float f) { return (bf16)f; }

// stage 16 consecutive f32 -> 16 bf16 into LDS (two v8bf stores)
static __device__ __forceinline__ void stage16_f32(const float* __restrict__ src,
                                                   bf16* dst) {
  const v8f* p = (const v8f*)src;  // src always 64B aligned here
  *(v8bf*)dst       = __builtin_convertvector(p[0], v8bf);
  *(v8bf*)(dst + 8) = __builtin_convertvector(p[1], v8bf);
}

static __device__ __forceinline__ v16bf cat8(v8bf lo, v8bf hi) {
  return __builtin_shufflevector(lo, hi, 0,1,2,3,4,5,6,7,8,9,10,11,12,13,14,15);
}

static __device__ __forceinline__ v8f wmma_bf16(v16bf a, v16bf b, v8f c) {
  // D = A(16x32 bf16) x B(32x16 bf16) + C(16x16 f32)
  return __builtin_amdgcn_wmma_f32_16x16x32_bf16(false, a, false, b, (short)0, c,
                                                 false, false);
}

#define LDA 40  // padded LDS row stride (elems); 80B keeps 16B alignment, skews banks

// A operand frag from LDS tile (rows x 32 k), contiguous k = 16*half + s (same order as ldsB)
static __device__ __forceinline__ v16bf ldsA(const bf16* base, int row, int half) {
  const bf16* p = base + row * LDA + 16 * half;
  return cat8(*(const v8bf*)p, *(const v8bf*)(p + 8));
}
// B operand frag: LDS holds W rows (n) x 32 k; same contiguous k order as ldsA:
// lane column n = lane%16, contiguous K range 16*(lane/16)..+16
static __device__ __forceinline__ v16bf ldsB(const bf16* base, int nrow, int half) {
  const bf16* p = base + nrow * LDA + 16 * half;
  return cat8(*(const v8bf*)p, *(const v8bf*)(p + 8));
}

#if HAVE_TDM
// TDM: DMA one 64-row x 32-col bf16 tile (row pitch = pitch elems) from global
// into LDS at lds_byte_addr, inserting 16B pad per 64B row (matches LDA=40).
static __device__ __forceinline__ void tdm_load_tile_bf16(const bf16* gsrc,
                                                          unsigned lds_byte_addr,
                                                          int pitch) {
  unsigned long long ga = (unsigned long long)(uintptr_t)gsrc;
  v4u g0;
  g0[0] = 1u;                                              // count=1 valid D#
  g0[1] = lds_byte_addr;                                   // lds_addr
  g0[2] = (unsigned)(ga & 0xffffffffu);                    // global_addr[31:0]
  g0[3] = (unsigned)((ga >> 32) & 0x01ffffffu) | (2u << 30); // ga[56:32], type=2
  v8i g1;
  // data_size=1 (2B), pad_enable, pad_interval=3 (16 DW = 64B row),
  // pad_amount=3 (4 DW = 16B)
  g1[0] = (1 << 16) | (1 << 20) | (3 << 22) | (3 << 25);
  g1[1] = (pitch & 0xffff) << 16;                          // tensor_dim0 lo16
  g1[2] = ((pitch >> 16) & 0xffff) | (64 << 16);           // dim0 hi16 | dim1 lo16
  g1[3] = (32 << 16);                                      // dim1 hi16 | tile_dim0=32
  g1[4] = 64;                                              // tile_dim1=64, tile_dim2=0
  g1[5] = pitch;                                           // tensor_dim0_stride lo32
  g1[6] = 0;                                               // stride hi | dim1_stride lo
  g1[7] = 0;
  v4i z4 = {};
#if __clang_major__ >= 23
  v8i z8 = {};
  __builtin_amdgcn_tensor_load_to_lds(g0, g1, z4, z4, z8, 0);
#else
  __builtin_amdgcn_tensor_load_to_lds(g0, g1, z4, z4, 0);
#endif
}
#endif

// ---------- kernel 1: fused QKV projection ----------
// grid (M/64, 3*NH), block 128 (4 waves). blockIdx.y selects {q,k,v} x head.
__global__ __launch_bounds__(128)
void mhsa_qkv_kernel(const float* __restrict__ x,
                     const float* __restrict__ Wq, const float* __restrict__ bq,
                     const float* __restrict__ Wk, const float* __restrict__ bk,
                     const float* __restrict__ Wv, const float* __restrict__ bv,
                     bf16* __restrict__ qws, bf16* __restrict__ kws,
                     bf16* __restrict__ vT) {
  __shared__ bf16 As[64 * LDA];
  __shared__ bf16 Bs[64 * LDA];

  const int tid  = threadIdx.x;
  const int w    = tid >> 5, lane = tid & 31;
  const int half = lane >> 4, col = lane & 15;
  const int m0   = blockIdx.x * 64;
  const int sel  = blockIdx.y / NH_;   // 0=q 1=k 2=v
  const int h    = blockIdx.y % NH_;
  const int nb   = h * 64;             // weight-row base

  const float* W    = (sel == 0) ? Wq : (sel == 1) ? Wk : Wv;
  const float* bias = (sel == 0) ? bq : (sel == 1) ? bk : bv;

  const int srow = tid >> 1;            // staging: 2 threads per row
  const int scol = (tid & 1) * 16;
  const float* xs = x + (size_t)(m0 + srow) * H_ + scol;
  const float* ws = W + (size_t)(nb + srow) * H_ + scol;

  v8f acc[4] = {};

  for (int kk = 0; kk < H_; kk += 32) {
    stage16_f32(xs + kk, As + srow * LDA + scol);
    stage16_f32(ws + kk, Bs + srow * LDA + scol);
    if (kk + 32 < H_) {  // pull next tiles toward the caches
      __builtin_prefetch(xs + kk + 32, 0, 1);
      __builtin_prefetch(ws + kk + 32, 0, 1);
    }
    __syncthreads();

    v16bf a  = ldsA(As, w * 16 + col, half);
    v16bf b0 = ldsB(Bs,  0 + col, half);
    v16bf b1 = ldsB(Bs, 16 + col, half);
    v16bf b2 = ldsB(Bs, 32 + col, half);
    v16bf b3 = ldsB(Bs, 48 + col, half);
    acc[0] = wmma_bf16(a, b0, acc[0]);
    acc[1] = wmma_bf16(a, b1, acc[1]);
    acc[2] = wmma_bf16(a, b2, acc[2]);
    acc[3] = wmma_bf16(a, b3, acc[3]);
    __syncthreads();
  }

  // epilogue: bias, scatter. C layout: elem r, lane -> row r+8*half, col 16t+lane%16
  for (int t = 0; t < 4; ++t) {
    const int d  = 16 * t + col;        // dim within head
    const float bv_ = bias[nb + d];
    for (int r = 0; r < 8; ++r) {
      const int m    = m0 + w * 16 + r + 8 * half;
      const int bidx = m >> 10, s = m & (S_ - 1);
      const int bh   = bidx * NH_ + h;
      float val = acc[t][r] + bv_;
      if (sel == 0) {        // q, pre-scaled by HD^-0.5
        qws[((size_t)bh * S_ + s) * HD_ + d] = f2bf1(val * 0.125f);
      } else if (sel == 1) { // k
        kws[((size_t)bh * S_ + s) * HD_ + d] = f2bf1(val);
      } else {               // v, stored transposed [b,h,d,s]
        vT[((size_t)bh * HD_ + d) * S_ + s] = f2bf1(val);
      }
    }
  }
}

// ---------- kernel 2: flash attention ----------
// grid (S/64, B*NH), block 128 (4 waves, 16 query rows each)
__global__ __launch_bounds__(128)
void mhsa_attn_kernel(const bf16* __restrict__ qws, const bf16* __restrict__ kws,
                      const bf16* __restrict__ vT, const int* __restrict__ amask,
                      bf16* __restrict__ ctxws) {
  __shared__ bf16 Pl[4][16 * 32];   // per-wave probs staging (C-layout -> A-layout)

  const int tid  = threadIdx.x;
  const int w    = tid >> 5, lane = tid & 31;
  const int half = lane >> 4, col = lane & 15;
  const int bh   = blockIdx.y;
  const int b    = bh / NH_, h = bh % NH_;
  const int q0   = blockIdx.x * 64 + w * 16;
  const int c0   = 16 * half;
  const float LOG2E = 1.4426950408889634f;

  // Q fragments: lane row = q0+col, dims split per A layout
  const bf16* qp = qws + ((size_t)bh * S_ + q0 + col) * HD_;
  const v16bf a0 = cat8(*(const v8bf*)(qp + c0),      *(const v8bf*)(qp + c0 + 8));
  const v16bf a1 = cat8(*(const v8bf*)(qp + c0 + 32), *(const v8bf*)(qp + c0 + 40));

  v8f ctx[4] = {};
  float mrun[8], lrun[8];
#pragma unroll
  for (int r = 0; r < 8; ++r) { mrun[r] = -1e30f; lrun[r] = 0.f; }

  bf16* myP = &Pl[w][0];
  const int* amb = amask + b * S_;
  const bf16* kbase = kws + ((size_t)bh * S_ + col) * HD_ + 16 * half;
  const bf16* vbase = vT + (size_t)bh * HD_ * S_ + (size_t)col * S_ + 16 * half;

  for (int kb = 0; kb < S_; kb += 32) {
    // ---- scores: two 16x16 tiles over 32 keys, K(head dim)=64 in two halves
    const bf16* kp0 = kbase + (size_t)kb * HD_;
    const bf16* kp1 = kp0 + 16 * HD_;
    v16bf k00 = cat8(*(const v8bf*)(kp0),      *(const v8bf*)(kp0 + 8));
    v16bf k01 = cat8(*(const v8bf*)(kp0 + 32), *(const v8bf*)(kp0 + 40));
    v16bf k10 = cat8(*(const v8bf*)(kp1),      *(const v8bf*)(kp1 + 8));
    v16bf k11 = cat8(*(const v8bf*)(kp1 + 32), *(const v8bf*)(kp1 + 40));
    if (kb + 32 < S_) __builtin_prefetch(kp0 + 32 * HD_, 0, 1);

    v8f z = {};
    v8f slo = wmma_bf16(a0, k00, z);
    slo     = wmma_bf16(a1, k01, slo);
    v8f shi = wmma_bf16(a0, k10, z);
    shi     = wmma_bf16(a1, k11, shi);

    // ---- mask (scale already folded into q)
    const int mv0 = amb[kb + col];
    const int mv1 = amb[kb + 16 + col];
    if (mv0 == 0) {
#pragma unroll
      for (int r = 0; r < 8; ++r) slo[r] = -10000.0f;
    }
    if (mv1 == 0) {
#pragma unroll
      for (int r = 0; r < 8; ++r) shi[r] = -10000.0f;
    }

    // ---- online softmax (rows striped: elem r, lane-half -> row r+8*half)
    float alpha[8], plo[8], phi[8];
#pragma unroll
    for (int r = 0; r < 8; ++r) {
      float bm = fmaxf(slo[r], shi[r]);
      bm = fmaxf(bm, __shfl_xor(bm, 1));
      bm = fmaxf(bm, __shfl_xor(bm, 2));
      bm = fmaxf(bm, __shfl_xor(bm, 4));
      bm = fmaxf(bm, __shfl_xor(bm, 8));
      const float nm = fmaxf(mrun[r], bm);
      alpha[r] = __builtin_amdgcn_exp2f((mrun[r] - nm) * LOG2E);
      plo[r]   = __builtin_amdgcn_exp2f((slo[r] - nm) * LOG2E);
      phi[r]   = __builtin_amdgcn_exp2f((shi[r] - nm) * LOG2E);
      float rs = plo[r] + phi[r];
      rs += __shfl_xor(rs, 1);
      rs += __shfl_xor(rs, 2);
      rs += __shfl_xor(rs, 4);
      rs += __shfl_xor(rs, 8);
      lrun[r] = lrun[r] * alpha[r] + rs;
      mrun[r] = nm;
    }

    // rescale running context
#pragma unroll
    for (int t = 0; t < 4; ++t)
#pragma unroll
      for (int r = 0; r < 8; ++r) ctx[t][r] *= alpha[r];

    // ---- re-layout P (C-layout -> A-layout) through wave-private LDS
#pragma unroll
    for (int r = 0; r < 8; ++r) {
      const int row = r + 8 * half;
      v2bf pr = pk2(plo[r], phi[r]);
      myP[row * 32 + col]      = pr[0];
      myP[row * 32 + 16 + col] = pr[1];
    }
    asm volatile("s_wait_dscnt 0" ::: "memory");  // cross-lane LDS RAW, same wave
    const v16bf pa = cat8(*(const v8bf*)(myP + col * 32 + c0),
                          *(const v8bf*)(myP + col * 32 + c0 + 8));

    // ---- ctx += P(16x32) x V(32x64); V^T gives contiguous keys per lane
    const bf16* vp = vbase + kb;
#pragma unroll
    for (int t = 0; t < 4; ++t) {
      const bf16* vpt = vp + (size_t)(16 * t) * S_;
      v16bf vb = cat8(*(const v8bf*)vpt, *(const v8bf*)(vpt + 8));
      ctx[t] = wmma_bf16(pa, vb, ctx[t]);
      if (kb + 32 < S_) __builtin_prefetch(vpt + 32, 0, 1);
    }
  }

  // ---- normalize and write ctx (bf16, [b, s, h*64+d]) for the output GEMM
  float inv[8];
#pragma unroll
  for (int r = 0; r < 8; ++r) inv[r] = 1.0f / lrun[r];
#pragma unroll
  for (int r = 0; r < 8; ++r) {
    const int s = q0 + r + 8 * half;
    bf16* dst = ctxws + ((size_t)b * S_ + s) * H_ + h * 64 + col;
    v2bf c01 = pk2(ctx[0][r] * inv[r], ctx[1][r] * inv[r]);
    v2bf c23 = pk2(ctx[2][r] * inv[r], ctx[3][r] * inv[r]);
    dst[0]  = c01[0];
    dst[16] = c01[1];
    dst[32] = c23[0];
    dst[48] = c23[1];
  }
}

// ---------- kernel 3: output projection ----------
// grid (M/64, H/64), block 128
__global__ __launch_bounds__(128)
void mhsa_proj_kernel(const bf16* __restrict__ ctxws, const float* __restrict__ Wo,
                      const float* __restrict__ bo, float* __restrict__ out) {
  __shared__ bf16 As[64 * LDA];
  __shared__ bf16 Bs[64 * LDA];

  const int tid  = threadIdx.x;
  const int w    = tid >> 5, lane = tid & 31;
  const int half = lane >> 4, col = lane & 15;
  const int m0   = blockIdx.x * 64;
  const int n0   = blockIdx.y * 64;
  const int srow = tid >> 1;
  const int scol = (tid & 1) * 16;
  const float* ws = Wo + (size_t)(n0 + srow) * H_ + scol;

#if HAVE_TDM
  const unsigned as_lds = (unsigned)(uintptr_t)(void*)As;
#endif

  v8f acc[4] = {};

  for (int kk = 0; kk < H_; kk += 32) {
#if HAVE_TDM
    if (w == 0) {  // one wave drives the Tensor Data Mover for the bf16 A tile
      tdm_load_tile_bf16(ctxws + (size_t)m0 * H_ + kk, as_lds, H_);
    }
#else
    {
      const v8bf* p = (const v8bf*)(ctxws + (size_t)(m0 + srow) * H_ + kk + scol);
      *(v8bf*)(As + srow * LDA + scol)     = p[0];
      *(v8bf*)(As + srow * LDA + scol + 8) = p[1];
    }
#endif
    stage16_f32(ws + kk, Bs + srow * LDA + scol);
    if (kk + 32 < H_) __builtin_prefetch(ws + kk + 32, 0, 1);
#if HAVE_TDM
    if (w == 0) __builtin_amdgcn_s_wait_tensorcnt(0);
#endif
    __syncthreads();

    v16bf a  = ldsA(As, w * 16 + col, half);
    v16bf b0 = ldsB(Bs,  0 + col, half);
    v16bf b1 = ldsB(Bs, 16 + col, half);
    v16bf b2 = ldsB(Bs, 32 + col, half);
    v16bf b3 = ldsB(Bs, 48 + col, half);
    acc[0] = wmma_bf16(a, b0, acc[0]);
    acc[1] = wmma_bf16(a, b1, acc[1]);
    acc[2] = wmma_bf16(a, b2, acc[2]);
    acc[3] = wmma_bf16(a, b3, acc[3]);
    __syncthreads();
  }

  for (int t = 0; t < 4; ++t) {
    const int n = n0 + 16 * t + col;
    const float bv_ = bo[n];
    for (int r = 0; r < 8; ++r) {
      const int m = m0 + w * 16 + r + 8 * half;
      out[(size_t)m * H_ + n] = acc[t][r] + bv_;
    }
  }
}

// ---------- launcher ----------
extern "C" void kernel_launch(void* const* d_in, const int* in_sizes, int n_in,
                              void* d_out, int out_size, void* d_ws, size_t ws_size,
                              hipStream_t stream) {
  const float* x  = (const float*)d_in[0];
  const int*   am = (const int*)  d_in[1];
  const float* Wq = (const float*)d_in[2];
  const float* bq = (const float*)d_in[3];
  const float* Wk = (const float*)d_in[4];
  const float* bk = (const float*)d_in[5];
  const float* Wv = (const float*)d_in[6];
  const float* bv = (const float*)d_in[7];
  const float* Wo = (const float*)d_in[8];
  const float* bo = (const float*)d_in[9];
  float* out = (float*)d_out;

  const size_t n1 = (size_t)B_ * NH_ * S_ * HD_;  // 6,291,456 elems per tensor
  bf16* qws = (bf16*)d_ws;
  bf16* kws = qws + n1;
  bf16* vT  = kws + n1;
  bf16* ctx = vT + n1;   // [B, S, H] bf16; total ws use ~50.3 MB

  mhsa_qkv_kernel<<<dim3(M_ / 64, 3 * NH_), 128, 0, stream>>>(
      x, Wq, bq, Wk, bk, Wv, bv, qws, kws, vT);
  mhsa_attn_kernel<<<dim3(S_ / 64, B_ * NH_), 128, 0, stream>>>(
      qws, kws, vT, am, ctx);
  mhsa_proj_kernel<<<dim3(M_ / 64, H_ / 64), 128, 0, stream>>>(
      ctx, Wo, bo, out);
}
